// BaseModel_70317204570750
// MI455X (gfx1250) — matrix operand order repaired and checked by reading, not compiled, not run on hardware
//
#include <hip/hip_runtime.h>
#include <cmath>

typedef __attribute__((ext_vector_type(16))) _Float16 v16h;
typedef __attribute__((ext_vector_type(8)))  float    v8f;

#define NE 34
#define NG 16
#define TOTF 1920
#define NITEMS_TP 120
#define NITEMS_MP 132

// ---- CG entry metadata (python-dict insertion order), constexpr for builders ----
constexpr int hE1[NE] = {0,0,0,0, 1,1,1,1,1,1,1,1,1, 2,2,2,2,2,2,2,2,2,2,2, 3,3,3,3,3,3,3,3,3,3};
constexpr int hE2[NE] = {0,1,2,3, 0,1,1,1,2,2,2,3,3, 0,1,1,1,2,2,2,2,3,3,3, 0,1,1,2,2,2,3,3,3,3};
constexpr int hEL[NE] = {0,1,2,3, 1,0,1,2,1,2,3,2,3, 2,1,2,3,0,1,2,3,1,2,3, 3,2,3,1,2,3,0,1,2,3};
constexpr int hCGO[NE] = {0,1,10,35, 84,93,102,129,174,219,294,399,504,
                          651,676,721,796,901,926,1001,1126,1301,1406,1581,
                          1826,1875,1980,2127,2232,2407,2652,2701,2848,3093};
constexpr int hKL[4] = {256,192,128,64};
constexpr int imax2(int a, int b) { return a > b ? a : b; }
constexpr int findCGO(int l1, int l2, int L) {
  for (int e = 0; e < NE; e++)
    if (hE1[e] == l1 && hE2[e] == l2 && hEL[e] == L) return hCGO[e];
  return 0;
}

// ---- (l1,l2) groups: all L-outputs of a pair stacked along WMMA M rows (<=16 rows) ----
struct GrpTab {
  unsigned char gl1[NG], gl2[NG], nrows[NG];
  short maxTP[NG], maxMP[NG];              // max kk over the group's rows
  unsigned char rowL[NG*16], rowM[NG*16], rowDL[NG*16];
  short rowCGO[NG*16], rowKKTP[NG*16], rowKKMP[NG*16];
};
constexpr GrpTab buildGrp() {
  GrpTab t{};
  for (int l1 = 0; l1 < 4; l1++)
    for (int l2 = 0; l2 < 4; l2++) {
      int gi = l1 * 4 + l2;
      t.gl1[gi] = (unsigned char)l1; t.gl2[gi] = (unsigned char)l2;
      int Lmin = l1 > l2 ? l1 - l2 : l2 - l1;
      int Lmax = (l1 + l2 > 3) ? 3 : (l1 + l2);
      int r = 0;
      for (int L = Lmin; L <= Lmax; L++) {
        int dL = 2 * L + 1, cgo = findCGO(l1, l2, L);
        int kktp = hKL[imax2(imax2(l1, l2), L)];
        int kkmp = hKL[imax2(l2, L)];
        for (int m = 0; m < dL; m++) {
          t.rowL[gi*16+r] = (unsigned char)L;
          t.rowM[gi*16+r] = (unsigned char)m;
          t.rowDL[gi*16+r] = (unsigned char)dL;
          t.rowCGO[gi*16+r] = (short)cgo;
          t.rowKKTP[gi*16+r] = (short)kktp;
          t.rowKKMP[gi*16+r] = (short)kkmp;
          r++;
        }
      }
      t.nrows[gi] = (unsigned char)r;
      t.maxTP[gi] = (short)hKL[imax2(imax2(l1, l2), Lmin)];
      t.maxMP[gi] = (short)hKL[imax2(l2, Lmin)];
    }
  return t;
}
__constant__ GrpTab cG = buildGrp();

// ---- constexpr-built schedules: item t -> (group, k-tile) ----
struct Items { unsigned char g[160]; unsigned char kt[160]; };
constexpr Items buildItems(bool mp) {
  Items it{};
  GrpTab gt = buildGrp();
  int idx = 0;
  for (int gi = 0; gi < NG; gi++) {
    int kk = mp ? gt.maxMP[gi] : gt.maxTP[gi];
    for (int t = 0; t < kk / 16; t++) { it.g[idx] = (unsigned char)gi; it.kt[idx] = (unsigned char)t; idx++; }
  }
  return it;
}
__constant__ Items cITP = buildItems(false);
__constant__ Items cIMP = buildItems(true);

// ---- (group, ab) -> (a, b) tables (avoid runtime div/mod) ----
struct ABTab { unsigned char a[NG * 64]; unsigned char b[NG * 64]; };
constexpr ABTab buildAB() {
  ABTab t{};
  for (int gi = 0; gi < NG; gi++) {
    int d2 = 2 * (gi & 3) + 1;
    for (int ab = 0; ab < 64; ab++) {
      t.a[gi * 64 + ab] = (unsigned char)(ab / d2);
      t.b[gi * 64 + ab] = (unsigned char)(ab % d2);
    }
  }
  return t;
}
__constant__ ABTab cAB = buildAB();

__constant__ int cKL[4]    = {256,192,128,64};
__constant__ int cACUM[4]  = {0,256,832,1472};   // per-atom l-block offsets (cumsum d_l*K_l)
__constant__ int cSHOFF[4] = {0,1,4,9};
__constant__ int cRBOFF[4] = {0,8,14,18};
__constant__ int cNML[4]   = {8,6,4,2};

// ---------------- CG constant generation: numpy RandomState(0).randn ----------------
__global__ void k_gen_cg(float* __restrict__ cg) {
  if (threadIdx.x != 0 || blockIdx.x != 0) return;
  unsigned mt[624]; int mti;
  mt[0] = 0u;                               // numpy mt19937_seed(0)
  for (int i = 1; i < 624; i++) mt[i] = 1812433253u * (mt[i-1] ^ (mt[i-1] >> 30)) + (unsigned)i;
  mti = 624;
  auto rand32 = [&]() -> unsigned {
    if (mti >= 624) {
      for (int k = 0; k < 624; k++) {
        unsigned y = (mt[k] & 0x80000000u) | (mt[(k+1)%624] & 0x7fffffffu);
        mt[k] = mt[(k+397)%624] ^ (y >> 1) ^ ((y & 1u) ? 0x9908b0dfu : 0u);
      }
      mti = 0;
    }
    unsigned y = mt[mti++];
    y ^= y >> 11; y ^= (y << 7) & 0x9d2c5680u; y ^= (y << 15) & 0xefc60000u; y ^= y >> 18;
    return y;
  };
  auto rdbl = [&]() -> double {
    unsigned a = rand32() >> 5, b = rand32() >> 6;
    return ((double)a * 67108864.0 + (double)b) / 9007199254740992.0;
  };
  double gcache = 0.0; bool hasg = false;
  auto gauss = [&]() -> double {
    if (hasg) { hasg = false; return gcache; }
    double x1, x2, r2;
    do {
      x1 = 2.0 * rdbl() - 1.0;
      x2 = 2.0 * rdbl() - 1.0;
      r2 = x1*x1 + x2*x2;
    } while (r2 >= 1.0 || r2 == 0.0);
    double f = sqrt(-2.0 * log(r2) / r2);
    gcache = f * x1; hasg = true;
    return f * x2;
  };
  int off = 0;
  for (int l1 = 0; l1 <= 3; l1++)
    for (int l2 = 0; l2 <= 3; l2++) {
      int lo = l1 > l2 ? l1 - l2 : l2 - l1;
      int hi = l1 + l2; if (hi > 3) hi = 3;
      for (int L = lo; L <= hi; L++) {
        int nv = (2*l1+1) * (2*l2+1) * (2*L+1);
        for (int v = 0; v < nv; v++) cg[off++] = (float)(gauss() * 0.2);
      }
    }
}

// -------- pre-pack stacked CG2 into WMMA A-fragment layout: [group][q][lane][16] f16 --------
__global__ void k_pack_afrag(const float* __restrict__ cg, _Float16* __restrict__ af) {
  int t = blockIdx.x * 256 + threadIdx.x;
  if (t >= NG * 2 * 32) return;
  int lane = t & 31;
  int q = (t >> 5) & 1;
  int gi = t >> 6;
  int l1 = cG.gl1[gi], l2 = cG.gl2[gi];
  int AB = (2*l1+1) * (2*l2+1);
  int nr = cG.nrows[gi];
  int g = lane >> 4, m = lane & 15;         // m = stacked row index
  _Float16* dst = af + (size_t)t * 16;
  #pragma unroll
  for (int i = 0; i < 16; i++) {
    int ab = (i & 7) + ((i >> 3) << 4) + (g << 3) + 32 * q;   // ISA A 16x32 f16 K-mapping
    float v = 0.f;
    if (m < nr && ab < AB)
      v = cg[(int)cG.rowCGO[gi*16+m] + ab * (int)cG.rowDL[gi*16+m] + (int)cG.rowM[gi*16+m]];
    dst[i] = (_Float16)v;
  }
}

// ---------------- per-atom species embedding ----------------
__global__ void k_embed(const float* __restrict__ tab, const int* __restrict__ spec,
                        float* __restrict__ emb, int N) {
  int t = blockIdx.x * 256 + threadIdx.x;
  if (t < N * 32) {
    int n = t >> 5, c = t & 31;
    emb[t] = tab[spec[n] * 32 + c];
  }
}

// ---------------- invariant message passing (spherical expansion) ----------------
__global__ __launch_bounds__(256)
void k_inv_mp(const float* __restrict__ sh, const float* __restrict__ rb,
              const float* __restrict__ emb, const int* __restrict__ ctr,
              const int* __restrict__ nbr, float* __restrict__ out, int N, int P) {
  int wave = threadIdx.x >> 5, lane = threadIdx.x & 31;
  int p = blockIdx.x * 8 + wave;
  if (p >= P) return;
  int ac = ctr[p];
  float ec = emb[(size_t)nbr[p] * 32 + lane];
  const float* shp = sh + (size_t)p * 16;
  const float* rbp = rb + (size_t)p * 20;
  for (int l = 0; l < 4; l++) {
    int d = 2*l+1, K = cKL[l], nr = cNML[l];
    float* ob = out + (size_t)N * cACUM[l];
    for (int m = 0; m < d; m++) {
      float s = 0.01f * shp[cSHOFF[l] + m];   // MP_SCALING * NU_SCALING
      float* row = ob + ((size_t)ac * d + m) * K + lane;
      for (int r = 0; r < nr; r++)
        atomicAdd(row + r * 32, s * rbp[cRBOFF[l] + r] * ec);
    }
  }
}

// ---------------- WMMA CG tensor product: out = in + 0.1*TP(in,in), one atom/block ----------------
__global__ __launch_bounds__(256)
void k_tp(const _Float16* __restrict__ afrag, const float* __restrict__ in,
          float* __restrict__ out, int N) {
  __shared__ float acc[TOTF];
  int tid = threadIdx.x, n = blockIdx.x;
  for (int i = tid; i < TOTF; i += 256) acc[i] = 0.f;
  __syncthreads();
  int wave = tid >> 5, lane = tid & 31, g = lane >> 4, col = lane & 15;
  const v16h* afp = (const v16h*)afrag;
  for (int t = wave; t < NITEMS_TP; t += 8) {
    int gi = cITP.g[t];
    int k0 = (int)cITP.kt[t] << 4;
    int l1 = cG.gl1[gi], l2 = cG.gl2[gi];
    int d1 = 2*l1+1, d2 = 2*l2+1, AB = d1 * d2;
    int nr = cG.nrows[gi];
    int K1 = cKL[l1], K2 = cKL[l2];
    const float* pa = in + (size_t)N * cACUM[l1] + ((size_t)n * d1) * K1 + k0;
    const float* pb = in + (size_t)N * cACUM[l2] + ((size_t)n * d2) * K2 + k0;
    v8f cacc = {};
    { // ab chunk 0
      v16h af = afp[(gi * 2 + 0) * 32 + lane];
      v16h bf;
      #pragma unroll
      for (int i = 0; i < 16; i++) {
        int ab = (i & 7) + ((i >> 3) << 4) + (g << 3);
        float v = 0.f;
        if (ab < AB) {
          int a = cAB.a[gi * 64 + ab], b = cAB.b[gi * 64 + ab];
          v = pa[a * K1 + col] * pb[b * K2 + col];
        }
        bf[i] = (_Float16)v;
      }
      cacc = __builtin_amdgcn_wmma_f32_16x16x32_f16(false, af, false, bf, (short)0, cacc, false, false);
    }
    if (AB > 32) { // ab chunk 1
      v16h af = afp[(gi * 2 + 1) * 32 + lane];
      v16h bf;
      #pragma unroll
      for (int i = 0; i < 16; i++) {
        int ab = 32 + (i & 7) + ((i >> 3) << 4) + (g << 3);
        float v = 0.f;
        if (ab < AB) {
          int a = cAB.a[gi * 64 + ab], b = cAB.b[gi * 64 + ab];
          v = pa[a * K1 + col] * pb[b * K2 + col];
        }
        bf[i] = (_Float16)v;
      }
      cacc = __builtin_amdgcn_wmma_f32_16x16x32_f16(false, af, false, bf, (short)0, cacc, false, false);
    }
    #pragma unroll
    for (int r = 0; r < 8; r++) {
      int row = r + 8 * g;
      if (row < nr && k0 < (int)cG.rowKKTP[gi*16+row]) {
        int L = cG.rowL[gi*16+row], m = cG.rowM[gi*16+row];
        atomicAdd(&acc[cACUM[L] + m * cKL[L] + k0 + col], cacc[r]);
      }
    }
  }
  __syncthreads();
  for (int i = tid; i < TOTF; i += 256) {
    int l = (i < 256) ? 0 : (i < 832) ? 1 : (i < 1472) ? 2 : 3;
    int loc = i - cACUM[l];
    int K = cKL[l], d = 2*l+1;
    int m = loc / K, k = loc % K;
    size_t gidx = (size_t)N * cACUM[l] + ((size_t)n * d + m) * K + k;
    out[gidx] = in[gidx] + 0.1f * acc[i];
  }
}

// ---------------- embed_centers: ef = feats * emb[n, k%32] ----------------
__global__ void k_embed_feats(const float* __restrict__ in, const float* __restrict__ emb,
                              float* __restrict__ out, int N) {
  size_t idx = (size_t)blockIdx.x * 256 + threadIdx.x;
  size_t tot = (size_t)N * TOTF;
  if (idx >= tot) return;
  size_t t1 = (size_t)N * 256, t2 = (size_t)N * 832, t3 = (size_t)N * 1472;
  int l; size_t base;
  if (idx < t1) { l = 0; base = 0; }
  else if (idx < t2) { l = 1; base = t1; }
  else if (idx < t3) { l = 2; base = t2; }
  else { l = 3; base = t3; }
  size_t loc = idx - base;
  int K = cKL[l], d = 2*l+1;
  int n = (int)(loc / (size_t)(d * K));
  int k = (int)(loc % (size_t)K);
  out[idx] = in[idx] * emb[(size_t)n * 32 + (k & 31)];
}

// ---------------- WMMA equivariant message passing, one pair/block ----------------
__global__ __launch_bounds__(256)
void k_mp(const _Float16* __restrict__ afrag, const float* __restrict__ sh,
          const float* __restrict__ rb, const float* __restrict__ ef,
          const int* __restrict__ ctr, const int* __restrict__ nbr,
          float* __restrict__ out, int N, int P) {
  __shared__ float acc[TOTF];
  int tid = threadIdx.x, p = blockIdx.x;
  for (int i = tid; i < TOTF; i += 256) acc[i] = 0.f;
  __syncthreads();
  int wave = tid >> 5, lane = tid & 31, g = lane >> 4, col = lane & 15;
  int an = nbr[p];
  const float* shp = sh + (size_t)p * 16;
  const float* rbp = rb + (size_t)p * 20;
  float rbs[4];
  for (int l = 0; l < 4; l++) {
    float s = 0.f;
    for (int r = 0; r < cNML[l]; r++) s += rbp[cRBOFF[l] + r];
    rbs[l] = s;
  }
  const v16h* afp = (const v16h*)afrag;
  for (int t = wave; t < NITEMS_MP; t += 8) {
    int gi = cIMP.g[t];
    int k0 = (int)cIMP.kt[t] << 4;
    int l1 = cG.gl1[gi], l2 = cG.gl2[gi];
    int d2 = 2*l2+1, AB = (2*l1+1) * d2;
    int nr = cG.nrows[gi];
    int K2 = cKL[l2];
    const float* pb = ef + (size_t)N * cACUM[l2] + ((size_t)an * d2) * K2 + k0;
    v8f cacc = {};
    { // ab chunk 0
      v16h af = afp[(gi * 2 + 0) * 32 + lane];
      v16h bf;
      #pragma unroll
      for (int i = 0; i < 16; i++) {
        int ab = (i & 7) + ((i >> 3) << 4) + (g << 3);
        float v = 0.f;
        if (ab < AB) {
          int a = cAB.a[gi * 64 + ab], b = cAB.b[gi * 64 + ab];
          float eA = 0.1f * shp[cSHOFF[l1] + a] * rbs[l1];   // nu-scaled sh * rb.sum
          v = eA * pb[b * K2 + col];
        }
        bf[i] = (_Float16)v;
      }
      cacc = __builtin_amdgcn_wmma_f32_16x16x32_f16(false, af, false, bf, (short)0, cacc, false, false);
    }
    if (AB > 32) { // ab chunk 1
      v16h af = afp[(gi * 2 + 1) * 32 + lane];
      v16h bf;
      #pragma unroll
      for (int i = 0; i < 16; i++) {
        int ab = 32 + (i & 7) + ((i >> 3) << 4) + (g << 3);
        float v = 0.f;
        if (ab < AB) {
          int a = cAB.a[gi * 64 + ab], b = cAB.b[gi * 64 + ab];
          float eA = 0.1f * shp[cSHOFF[l1] + a] * rbs[l1];
          v = eA * pb[b * K2 + col];
        }
        bf[i] = (_Float16)v;
      }
      cacc = __builtin_amdgcn_wmma_f32_16x16x32_f16(false, af, false, bf, (short)0, cacc, false, false);
    }
    #pragma unroll
    for (int r = 0; r < 8; r++) {
      int row = r + 8 * g;
      if (row < nr && k0 < (int)cG.rowKKMP[gi*16+row]) {
        int L = cG.rowL[gi*16+row], m = cG.rowM[gi*16+row];
        atomicAdd(&acc[cACUM[L] + m * cKL[L] + k0 + col], cacc[r]);
      }
    }
  }
  __syncthreads();
  int ac = ctr[p];
  for (int i = tid; i < TOTF; i += 256) {
    float v = acc[i];
    if (v != 0.f) {
      int l = (i < 256) ? 0 : (i < 832) ? 1 : (i < 1472) ? 2 : 3;
      int loc = i - cACUM[l];
      int K = cKL[l], d = 2*l+1;
      int m = loc / K, k = loc % K;
      atomicAdd(&out[(size_t)N * cACUM[l] + ((size_t)ac * d + m) * K + k], 0.1f * v);  // MP_SCALING
    }
  }
}

// ---------------- energy head: out[n] = b + sum_k f0[n,k]*emb[n,k%32]*w[k] ----------------
__global__ __launch_bounds__(256)
void k_energy(const float* __restrict__ f0, const float* __restrict__ emb,
              const float* __restrict__ w, const float* __restrict__ b,
              float* __restrict__ out, int N) {
  int wave = threadIdx.x >> 5, lane = threadIdx.x & 31;
  int n = blockIdx.x * 8 + wave;
  if (n >= N) return;
  float e = emb[(size_t)n * 32 + lane];
  const float* fr = f0 + (size_t)n * 256;
  float s = 0.f;
  #pragma unroll
  for (int j = 0; j < 8; j++) {
    int k = lane + j * 32;
    s += fr[k] * e * w[k];
  }
  #pragma unroll
  for (int o = 16; o > 0; o >>= 1) s += __shfl_down(s, o, 32);
  if (lane == 0) out[n] = s + b[0];
}

extern "C" void kernel_launch(void* const* d_in, const int* in_sizes, int n_in,
                              void* d_out, int out_size, void* d_ws, size_t ws_size,
                              hipStream_t stream) {
  const float* sh   = (const float*)d_in[0];
  const float* rb   = (const float*)d_in[1];
  const float* embt = (const float*)d_in[2];
  const float* wE   = (const float*)d_in[3];
  const float* bE   = (const float*)d_in[4];
  const int*   spec = (const int*)d_in[5];
  const int*   ctr  = (const int*)d_in[6];
  const int*   nbr  = (const int*)d_in[7];
  const int N = in_sizes[5];
  const int P = in_sizes[6];

  float* ws  = (float*)d_ws;
  float*     cg    = ws;                                  // 4096 floats (3436 used)
  _Float16*  afrag = (_Float16*)(ws + 4096);              // NG*2*32*16 f16 = 8192 float-equivs
  float*     emb   = ws + 4096 + 8192;                    // N*32
  float*     B0    = emb + (size_t)N * 32;                // N*1920
  float*     B1    = B0 + (size_t)N * TOTF;               // N*1920
  float*     B2    = B1 + (size_t)N * TOTF;               // N*1920
  size_t need = ((size_t)4096 + 8192 + (size_t)N * 32 + 3u * (size_t)N * TOTF) * sizeof(float);
  if (ws_size < need) return;

  k_gen_cg<<<1, 1, 0, stream>>>(cg);
  k_pack_afrag<<<(NG * 64 + 255) / 256, 256, 0, stream>>>(cg, afrag);
  k_embed<<<(N * 32 + 255) / 256, 256, 0, stream>>>(embt, spec, emb, N);

  hipMemsetAsync(B0, 0, (size_t)N * TOTF * sizeof(float), stream);
  k_inv_mp<<<(P + 7) / 8, 256, 0, stream>>>(sh, rb, emb, ctr, nbr, B0, N, P);

  k_tp<<<N, 256, 0, stream>>>(afrag, B0, B1, N);          // feats after 1st CG iterate

  k_embed_feats<<<(int)(((size_t)N * TOTF + 255) / 256), 256, 0, stream>>>(B1, emb, B0, N);  // ef

  hipMemsetAsync(B2, 0, (size_t)N * TOTF * sizeof(float), stream);
  k_mp<<<P, 256, 0, stream>>>(afrag, sh, rb, B0, ctr, nbr, B2, N, P);

  k_tp<<<N, 256, 0, stream>>>(afrag, B2, B1, N);          // feats after 2nd CG iterate

  k_energy<<<(N + 7) / 8, 256, 0, stream>>>(B1, emb, wE, bE, (float*)d_out, N);
}
